// LinearProjection_48576080118602
// MI455X (gfx1250) — compile-verified, matrix operand order checked by reading, not run on hardware
//
#include <hip/hip_runtime.h>
#include <hip/hip_bf16.h>
#include <stdint.h>
#include <stddef.h>

// ---------------------------------------------------------------------------
// Masked linear projection as a bf16 WMMA GEMM (MI455X / gfx1250, wave32).
//   M = 32*512 = 16384 rows, K = 3133 (padded to 3136 = 98*32), N = 256.
// Memory-bound: ~220 MB -> ~9.4 us @ 23.3 TB/s; bf16 WMMA keeps matrix-core
// time under that floor. Pipelined: double-buffered LDS A tile, register
// prefetch issued after B-frag loads (finer s_wait_loadcnt), 1 barrier/iter,
// branch-free hot loop (tail iterations peeled). Grid is N-major so the 4
// blocks sharing an A slab are dispatch-adjacent -> A read from HBM once.
// ---------------------------------------------------------------------------

typedef __attribute__((ext_vector_type(16))) __bf16 v16bf;
typedef __attribute__((ext_vector_type(8)))  __bf16 v8bf;
typedef __attribute__((ext_vector_type(4)))  __bf16 v4bf;
typedef __attribute__((ext_vector_type(8)))  float  v8f;
typedef __attribute__((ext_vector_type(4)))  float  v4f;

#define TOKDIM 256
#define FEAT   3133
#define NKB    98         // K blocks of 32 (3136 padded)
#define NTB    16         // N blocks of 16
#define MROWS  16384      // B*N
#define EMB    3072
#define TAILW  64         // 61 tail feats + 3 zero pad

#define WB_BYTES ((size_t)NTB * NKB * 32 * 16 * 2)   // 1,605,632 B of packed W

#define BM 128
#define BN 64
#define LDSTR 40          // bf16 elems per LDS A row (80 B, aligned b128 reads)

// Pack W[t][k] (fp32) into bf16 B-fragments in WMMA lane order.
__global__ void pack_w_kernel(const float* __restrict__ W, __bf16* __restrict__ Wb) {
  int id = blockIdx.x * blockDim.x + threadIdx.x;      // (tb*NKB + kb)*32 + lane
  if (id >= NTB * NKB * 32) return;
  int lane = id & 31;
  int tile = id >> 5;
  int kb   = tile % NKB;
  int tb   = tile / NKB;
  int t    = tb * 16 + (lane & 15);
  int hi   = lane >> 4;
  __bf16* dst = Wb + (size_t)id * 16;
#pragma unroll
  for (int j = 0; j < 16; ++j) {
    int klocal = (j < 8) ? (hi * 8 + j) : (16 + hi * 8 + (j - 8));
    int k = kb * 32 + klocal;
    float v = (k < FEAT) ? W[(size_t)t * FEAT + k] : 0.0f;
    dst[j] = (__bf16)v;
  }
}

// Pack [vis(6) | bbox(4) | kpts(51) | 0,0,0] per row into bf16 [MROWS][64].
__global__ void pack_tail_kernel(const float* __restrict__ vis,
                                 const float* __restrict__ bbox,
                                 const float* __restrict__ kpt,
                                 __bf16* __restrict__ tail) {
  int id = blockIdx.x * blockDim.x + threadIdx.x;
  if (id >= MROWS * TAILW) return;
  int j   = id & 63;
  int row = id >> 6;
  float v = 0.0f;
  if (j < 6)        v = vis[row * 6 + j];
  else if (j < 10)  v = bbox[row * 4 + (j - 6)];
  else if (j < 61)  v = kpt[row * 51 + (j - 10)];
  tail[id] = (__bf16)v;
}

__global__ __launch_bounds__(256)
void gemm_bf16_wmma_kernel(const float*  __restrict__ emb,
                           const __bf16* __restrict__ tail,
                           const __bf16* __restrict__ Wb,
                           const float*  __restrict__ bias,
                           const uint8_t* __restrict__ mask,
                           float* __restrict__ out) {
  __shared__ __bf16 As[2][BM * LDSTR];                 // 2 x 10,240 B

  const int tid   = threadIdx.x;
  const int lane  = tid & 31;
  const int wave  = tid >> 5;                          // 8 waves
  const int waveM = wave & 3;                          // 4 in M
  const int waveN = wave >> 2;                         // 2 in N
  const int hi    = lane >> 4;
  const int m16   = lane & 15;
  const int colBase = blockIdx.x * BN;                 // 4 N-tiles  (fastest)
  const int rowBase = blockIdx.y * BM;                 // 128 M-tiles
  const int tb0     = (colBase >> 4) + waveN * 2;

  v8f acc[2][2] = {};

  // cooperative A-tile loader indexing: 256 threads, 128 rows x 8 chunks of 4
  const int lr = tid >> 3;                             // row 0..31 (+32*i)
  const int lc = tid & 7;                              // chunk 0..7

  v4f  pf[4];                                          // fp32 prefetch regs
  v4bf pb[4];                                          // bf16 prefetch regs
  v16bf afrag[2], bfrag[2];

  // --- pipeline helper lambdas (inlined by the compiler) ---
  auto issue_emb = [&](int nkb) {
    const float* src = emb + (size_t)rowBase * EMB + (size_t)nkb * 32 + lc * 4;
#pragma unroll
    for (int i = 0; i < 4; ++i)
      pf[i] = *(const v4f*)(src + (size_t)(lr + i * 32) * EMB);
  };
  auto drain_emb = [&](__bf16* buf) {
#pragma unroll
    for (int i = 0; i < 4; ++i) {
      __bf16* dst = &buf[(lr + i * 32) * LDSTR + lc * 4];
      dst[0] = (__bf16)pf[i].x; dst[1] = (__bf16)pf[i].y;
      dst[2] = (__bf16)pf[i].z; dst[3] = (__bf16)pf[i].w;
    }
  };
  auto issue_tail = [&](int nkb) {
    const __bf16* src = tail + (size_t)rowBase * TAILW + (nkb - 96) * 32 + lc * 4;
#pragma unroll
    for (int i = 0; i < 4; ++i)
      pb[i] = *(const v4bf*)(src + (size_t)(lr + i * 32) * TAILW);
  };
  auto drain_tail = [&](__bf16* buf) {
#pragma unroll
    for (int i = 0; i < 4; ++i)
      *(v4bf*)&buf[(lr + i * 32) * LDSTR + lc * 4] = pb[i];
  };
  // B frags (L2) + A frags (LDS) issued BEFORE the A prefetch so the WMMA
  // wait only covers these loads, leaving the prefetch in flight.
  auto load_frags = [&](int kb, const __bf16* buf) {
#pragma unroll
    for (int fn = 0; fn < 2; ++fn)
      bfrag[fn] = *(const v16bf*)(Wb +
                    ((size_t)((tb0 + fn) * NKB + kb) * 32 + lane) * 16);
#pragma unroll
    for (int fm = 0; fm < 2; ++fm) {
      const __bf16* base = buf + (waveM * 32 + fm * 16 + m16) * LDSTR + hi * 8;
      v8bf lo = *(const v8bf*)base;                    // k = hi*8 ..
      v8bf h8 = *(const v8bf*)(base + 16);             // k = 16+hi*8 ..
      afrag[fm] = __builtin_shufflevector(lo, h8,
                    0,1,2,3,4,5,6,7,8,9,10,11,12,13,14,15);
    }
  };
  auto do_wmma = [&]() {
#pragma unroll
    for (int fm = 0; fm < 2; ++fm)
#pragma unroll
      for (int fn = 0; fn < 2; ++fn)
        acc[fm][fn] = __builtin_amdgcn_wmma_f32_16x16x32_bf16(
            false, afrag[fm], false, bfrag[fn],
            (short)0, acc[fm][fn], false, false);
  };

  // ---- prologue: stage tile kb=0 ----
  issue_emb(0);
  drain_emb(As[0]);
  __syncthreads();

  // ---- branch-free hot loop: kb = 0..94, next tile is always embeddings ----
  for (int kb = 0; kb < 95; ++kb) {
    load_frags(kb, As[kb & 1]);
    issue_emb(kb + 1);
    do_wmma();
    drain_emb(As[(kb + 1) & 1]);
    __syncthreads();
  }
  // ---- peeled: kb = 95 (prefetch tail 96), 96 (prefetch tail 97), 97 ----
  load_frags(95, As[1]);
  issue_tail(96);
  do_wmma();
  drain_tail(As[0]);
  __syncthreads();

  load_frags(96, As[0]);
  issue_tail(97);
  do_wmma();
  drain_tail(As[1]);
  __syncthreads();

  load_frags(97, As[1]);
  do_wmma();

  // ---- epilogue: + bias, apply row mask (masked rows -> 0) ----
#pragma unroll
  for (int fm = 0; fm < 2; ++fm) {
    int row0 = rowBase + waveM * 32 + fm * 16 + hi * 8;
#pragma unroll
    for (int fn = 0; fn < 2; ++fn) {
      int col = colBase + waveN * 32 + fn * 16 + m16;
      float bv = bias[col];
#pragma unroll
      for (int i = 0; i < 8; ++i) {
        int row = row0 + i;
        float v = mask[row] ? (acc[fm][fn][i] + bv) : 0.0f;
        out[(size_t)row * TOKDIM + col] = v;
      }
    }
  }
}

extern "C" void kernel_launch(void* const* d_in, const int* in_sizes, int n_in,
                              void* d_out, int out_size, void* d_ws, size_t ws_size,
                              hipStream_t stream) {
  const float*   emb  = (const float*)d_in[0];
  const float*   vis  = (const float*)d_in[1];
  const float*   bbox = (const float*)d_in[2];
  const float*   kpt  = (const float*)d_in[3];
  const uint8_t* mask = (const uint8_t*)d_in[4];   // jnp.bool_ -> 1 byte/elem
  const float*   W    = (const float*)d_in[5];
  const float*   bias = (const float*)d_in[6];
  float*         out  = (float*)d_out;

  __bf16* Wb   = (__bf16*)d_ws;
  __bf16* tail = (__bf16*)((char*)d_ws + WB_BYTES);

  pack_w_kernel<<<(NTB * NKB * 32 + 255) / 256, 256, 0, stream>>>(W, Wb);
  pack_tail_kernel<<<(MROWS * TAILW) / 256, 256, 0, stream>>>(vis, bbox, kpt, tail);

  // N-major grid: sibling blocks (same A slab) are adjacent in dispatch order
  dim3 grid(TOKDIM / BN, MROWS / BM);                // 4 x 128
  gemm_bf16_wmma_kernel<<<grid, 256, 0, stream>>>(emb, tail, Wb, bias, mask, out);
}
